// ValueNetwork_82789789598294
// MI455X (gfx1250) — matrix operand order repaired and checked by reading, not compiled, not run on hardware
//
#include <hip/hip_runtime.h>
#include <hip/hip_bf16.h>
#include <math.h>

typedef _Float16 h16;
typedef __attribute__((ext_vector_type(16))) _Float16 v16h;
typedef __attribute__((ext_vector_type(8)))  float    v8f;

// Padded row geometry for pair / conv1 activation buffers:
//   row stride 132 halfs; element w lives at row[2+w]; row[0..1] and
//   row[130..131] are zero halos so 3x3 SAME conv B-fragments need no
//   bounds checks (w index range used: -1 .. 128).
#define RS 132

// ---------------------------------------------------------------------------
// K0: pre-pack conv weights as f16 WMMA A-fragments (ISA 7.12.2 layout):
//   lane -> M = lane&15, g = lane>>4 ; element e -> K = 8g + (e&7) + 16(e>>3)
//   wpack1[tap][ot(4)][lane][e]        (conv1: 64oc x 32c, 9 taps)
//   wpack2[tap][kk(2)][ot(2)][lane][e] (conv2: 32oc x 64c, 9 taps)
// ---------------------------------------------------------------------------
__global__ __launch_bounds__(256) void k_prep(
    const float* __restrict__ w_c1, const float* __restrict__ w_c2,
    h16* __restrict__ wpack1, h16* __restrict__ wpack2)
{
    int idx = blockIdx.x * blockDim.x + threadIdx.x;   // 36864 total
    if (idx < 18432) {
        int e = idx & 15, lane = (idx >> 4) & 31, ot = (idx >> 9) & 3, tap = idx >> 11;
        int m = lane & 15, g = lane >> 4;
        int k = (g << 3) + (e & 7) + ((e >> 3) << 4);
        wpack1[idx] = (h16)w_c1[(((ot * 16 + m) << 5) + k) * 9 + tap];
    } else {
        int j = idx - 18432;
        int e = j & 15, lane = (j >> 4) & 31, ot = (j >> 9) & 1, kk = (j >> 10) & 1, tap = j >> 11;
        int m = lane & 15, g = lane >> 4;
        int k = (kk << 5) + (g << 3) + (e & 7) + ((e >> 3) << 4);
        wpack2[j] = (h16)w_c2[(((ot * 16 + m) << 6) + k) * 9 + tap];
    }
}

// ---------------------------------------------------------------------------
// K1: front MLP  agents(7) -> relu(64) -> relu(32); writes in_out (B,N,32)
//     and transposed copy fT (B,32,N).
// ---------------------------------------------------------------------------
__global__ __launch_bounds__(256) void k_front(
    const float* __restrict__ state,
    const float* __restrict__ w_in1, const float* __restrict__ b_in1,
    const float* __restrict__ w_in2, const float* __restrict__ b_in2,
    float* __restrict__ in_out, float* __restrict__ fT)
{
    int idx = blockIdx.x * blockDim.x + threadIdx.x;   // (b,n), 8192 total
    if (idx >= 64 * 128) return;
    int b = idx >> 7, n = idx & 127;
    const float* ag = state + idx * 13 + 6;
    float a[7];
#pragma unroll
    for (int k = 0; k < 7; ++k) a[k] = ag[k];
    float h[64];
#pragma unroll
    for (int o = 0; o < 64; ++o) {
        float acc = b_in1[o];
#pragma unroll
        for (int k = 0; k < 7; ++k) acc += w_in1[o * 7 + k] * a[k];
        h[o] = acc > 0.f ? acc : 0.f;
    }
#pragma unroll
    for (int c = 0; c < 32; ++c) {
        float acc = b_in2[c];
#pragma unroll
        for (int o = 0; o < 64; ++o) acc += w_in2[c * 64 + o] * h[o];
        acc = acc > 0.f ? acc : 0.f;
        in_out[idx * 32 + c] = acc;
        fT[((b * 32 + c) << 7) + n] = acc;
    }
}

// ---------------------------------------------------------------------------
// K2: pair[b,c,i,j] = f_i - f_j + f_i*(i==j), f16, halo-padded rows.
// One 128-thread block per (b,c,i) row; writes all 132 slots (halos = 0).
// ---------------------------------------------------------------------------
__global__ __launch_bounds__(128) void k_pair(
    const float* __restrict__ fT, h16* __restrict__ pairP)
{
    int row = blockIdx.x;                       // (b*32+c)*128 + h, 262144 rows
    int bc = row >> 7, h = row & 127;
    const float* f = fT + ((long)bc << 7);
    float fh = f[h];
    h16* rp = pairP + (long)row * RS;
    for (int wp = threadIdx.x; wp < RS; wp += 128) {
        int w = wp - 2;
        bool inside = (w >= 0) & (w < 128);
        float fw = f[inside ? w : 0];
        float v = inside ? (fh - fw + (h == w ? fh : 0.f)) : 0.f;
        rp[wp] = (h16)v;
    }
}

// ---------------------------------------------------------------------------
// K3: conv1 3x3 SAME, 32->64ch, implicit GEMM. One wave = 64oc x 16w tile:
//     per tap, ONE B fragment feeds 4 WMMAs (4 oc-tiles). 9 taps x 4 = 36 WMMA.
// ---------------------------------------------------------------------------
__global__ __launch_bounds__(256) void k_conv1(
    const h16* __restrict__ pairP, const h16* __restrict__ wpack1,
    const float* __restrict__ b_c1, h16* __restrict__ outP)
{
    int wid  = (blockIdx.x * blockDim.x + threadIdx.x) >> 5;   // 65536 waves
    int lane = threadIdx.x & 31;
    int wt = wid & 7, h = (wid >> 3) & 127, b = wid >> 10;
    int w0 = wt << 4;
    int g = lane >> 4;
    const v16h* wp = (const v16h*)wpack1;

    v8f acc[4];
#pragma unroll
    for (int ot = 0; ot < 4; ++ot) acc[ot] = (v8f){0.f,0.f,0.f,0.f,0.f,0.f,0.f,0.f};

#pragma unroll
    for (int kh = 0; kh < 3; ++kh) {
        int hh = h + kh - 1;
        if (hh < 0 || hh >= 128) continue;                     // wave-uniform
#pragma unroll
        for (int kw = 0; kw < 3; ++kw) {
            v16h Bv;
#pragma unroll
            for (int s = 0; s < 16; ++s) {
                int kc = (g << 3) + (s & 7) + ((s >> 3) << 4);
                const h16* rp = pairP + ((((long)b << 5) + kc) * 128 + hh) * RS
                              + 2 + (w0 + (lane & 15) + kw - 1);
                Bv[s] = rp[0];                                 // halo-safe
            }
            int tap = kh * 3 + kw;
#pragma unroll
            for (int ot = 0; ot < 4; ++ot) {
                v16h A = wp[(tap * 4 + ot) * 32 + lane];
                acc[ot] = __builtin_amdgcn_wmma_f32_16x16x32_f16(
                              false, A, false, Bv, (short)0, acc[ot], false, false);
            }
        }
    }
#pragma unroll
    for (int ot = 0; ot < 4; ++ot) {
#pragma unroll
        for (int r = 0; r < 8; ++r) {
            int oc = (ot << 4) + r + (g << 3);
            float v = acc[ot][r] + b_c1[oc];
            v = v > 0.f ? v : 0.f;                             // fused ReLU
            h16* orp = outP + ((((long)b << 6) + oc) * 128 + h) * RS;
            orp[2 + w0 + (lane & 15)] = (h16)v;
            if (wt == 0 && (lane & 15) < 2)  orp[lane & 15] = (h16)0.f;        // left halo
            if (wt == 7 && (lane & 15) >= 14) orp[116 + (lane & 15)] = (h16)0.f; // right halo
        }
    }
}

// ---------------------------------------------------------------------------
// K4: conv2 (64->32) + bias + softmax over w + afa, fused per (b,h) row.
//     8 waves (256 thr): wave = w-tile; each wave does all 32 oc.
//     Per tap/kk: ONE B fragment feeds 2 WMMAs. 9 x 2 x 2 = 36 static WMMAs.
// ---------------------------------------------------------------------------
__global__ __launch_bounds__(256) void k_conv2_softmax_afa(
    const h16* __restrict__ c1P, const h16* __restrict__ pairP,
    const h16* __restrict__ wpack2, const float* __restrict__ b_c2,
    float* __restrict__ afa)
{
    __shared__ float row[32][128];     // conv2 output
    __shared__ h16   prow[32][128];    // pair row slab for afa
    __shared__ float pred[3][32][8];   // partial max / sum / afa
    __shared__ float rmax[32];

    int tid = threadIdx.x;
    int b = blockIdx.x >> 7, h = blockIdx.x & 127;
    const v16h* wp = (const v16h*)wpack2;

    for (int i = tid; i < 32 * 128; i += 256)
        prow[i >> 7][i & 127] =
            pairP[((((long)b << 5) + (i >> 7)) * 128 + h) * RS + 2 + (i & 127)];

    int wt = tid >> 5, lane = tid & 31;
    int w0 = wt << 4;
    int g = lane >> 4;

    v8f acc[2];
#pragma unroll
    for (int ot = 0; ot < 2; ++ot) acc[ot] = (v8f){0.f,0.f,0.f,0.f,0.f,0.f,0.f,0.f};

#pragma unroll
    for (int kh = 0; kh < 3; ++kh) {
        int hh = h + kh - 1;
        if (hh < 0 || hh >= 128) continue;
#pragma unroll
        for (int kw = 0; kw < 3; ++kw) {
            int tap = kh * 3 + kw;
#pragma unroll
            for (int kk = 0; kk < 2; ++kk) {                   // K = 64
                v16h Bv;
#pragma unroll
                for (int s = 0; s < 16; ++s) {
                    int c = (kk << 5) + (g << 3) + (s & 7) + ((s >> 3) << 4);
                    const h16* rp = c1P + ((((long)b << 6) + c) * 128 + hh) * RS
                                  + 2 + (w0 + (lane & 15) + kw - 1);
                    Bv[s] = rp[0];
                }
#pragma unroll
                for (int ot = 0; ot < 2; ++ot) {
                    v16h A = wp[((tap * 2 + kk) * 2 + ot) * 32 + lane];
                    acc[ot] = __builtin_amdgcn_wmma_f32_16x16x32_f16(
                                  false, A, false, Bv, (short)0, acc[ot], false, false);
                }
            }
        }
    }
#pragma unroll
    for (int ot = 0; ot < 2; ++ot) {
#pragma unroll
        for (int r = 0; r < 8; ++r) {
            int oc = (ot << 4) + r + (g << 3);
            row[oc][w0 + (lane & 15)] = acc[ot][r] + b_c2[oc];
        }
    }
    __syncthreads();

    // softmax over w (128) per oc row + afa reduction; 8 threads per row.
    int co = tid >> 3, s = tid & 7;
    float mx = -3.0e38f;
    for (int j = 0; j < 16; ++j) { float v = row[co][s * 16 + j]; mx = v > mx ? v : mx; }
    pred[0][co][s] = mx;
    __syncthreads();
    if (tid < 32) {
        float v = pred[0][tid][0];
        for (int t = 1; t < 8; ++t) { float u = pred[0][tid][t]; v = u > v ? u : v; }
        rmax[tid] = v;
    }
    __syncthreads();
    float sm = 0.f, sa = 0.f;
    for (int j = 0; j < 16; ++j) {
        int w = s * 16 + j;
        float e = __expf(row[co][w] - rmax[co]);
        sm += e;
        sa += e * (float)prow[co][w];
    }
    pred[1][co][s] = sm; pred[2][co][s] = sa;
    __syncthreads();
    if (tid < 32) {
        float ssum = 0.f, asum = 0.f;
        for (int t = 0; t < 8; ++t) { ssum += pred[1][tid][t]; asum += pred[2][tid][t]; }
        afa[b * 4096 + tid * 128 + h] = asum / ssum;  // flat (C,N) layout per b
    }
}

// ---------------------------------------------------------------------------
// K5: everything after afa, one workgroup (128 threads = one per n) per batch.
//     All reductions deterministic (serial sums, no float atomics).
// ---------------------------------------------------------------------------
__global__ __launch_bounds__(128) void k_tail(
    const float* __restrict__ state, const float* __restrict__ in_out,
    const float* __restrict__ afa,
    const float* __restrict__ w_ia1, const float* __restrict__ b_ia1,
    const float* __restrict__ w_ia2, const float* __restrict__ b_ia2,
    const float* __restrict__ w_s1,  const float* __restrict__ b_s1,
    const float* __restrict__ w_s2,  const float* __restrict__ b_s2,
    const float* __restrict__ w_a1,  const float* __restrict__ b_a1,
    const float* __restrict__ w_a2,  const float* __restrict__ b_a2,
    const float* __restrict__ w_a3,  const float* __restrict__ b_a3,
    const float* __restrict__ w_z, const float* __restrict__ w_r,
    const float* __restrict__ w_q,
    const float* __restrict__ w_m1, const float* __restrict__ b_m1,
    const float* __restrict__ w_m2, const float* __restrict__ b_m2,
    const float* __restrict__ w_m3, const float* __restrict__ b_m3,
    float* __restrict__ out)
{
    __shared__ float smi[128][64];    // sort_mlp_input rows
    __shared__ float wgt[128][64];    // weights * smi  (aliased as gru_input)
    __shared__ float t64[128][64];    // h1 / s1 / a1, then r (GRU reset gate)
    __shared__ float t38[128][38];    // x / sort_in / sort_out / a2 scratch
    __shared__ float hnm[128][64];    // per-column hn before mean
    __shared__ float gsum[32];
    __shared__ float hnacc[64];
    __shared__ float sebuf[128];
    __shared__ float m1b[128];
    __shared__ float m2b[64];
    __shared__ float ssum;

    int b = blockIdx.x, n = threadIdx.x;

    // ---- per-n MLP chain: x -> ia -> sort_out ----
    const float* io = in_out + (b * 128 + n) * 32;
    const float* af = afa + b * 4096 + n * 32;     // flat reshape (C,N)->(N,C)
    for (int c = 0; c < 32; ++c) t38[n][c] = io[c] + af[c];
    for (int o = 0; o < 64; ++o) {
        float acc = b_ia1[o];
        for (int j = 0; j < 32; ++j) acc += w_ia1[o * 32 + j] * t38[n][j];
        t64[n][o] = acc > 0.f ? acc : 0.f;
    }
    for (int o = 0; o < 32; ++o) {
        float acc = b_ia2[o];
        for (int j = 0; j < 64; ++j) acc += w_ia2[o * 64 + j] * t64[n][j];
        t38[n][o] = acc;                            // ia (overwrites x)
    }
    for (int j = 0; j < 6; ++j) t38[n][32 + j] = state[(b * 128 + n) * 13 + j];
    for (int o = 0; o < 64; ++o) {
        float acc = b_s1[o];
        for (int j = 0; j < 38; ++j) acc += w_s1[o * 38 + j] * t38[n][j];
        t64[n][o] = acc > 0.f ? acc : 0.f;
    }
    for (int o = 0; o < 32; ++o) {
        float acc = b_s2[o];
        for (int j = 0; j < 64; ++j) acc += w_s2[o * 64 + j] * t64[n][j];
        t38[n][o] = acc;                            // sort_out
    }
    __syncthreads();
    if (n < 32) {                                   // gstate = mean_n(sort_out)
        float s = 0.f;
        for (int j = 0; j < 128; ++j) s += t38[j][n];
        gsum[n] = s * (1.f / 128.f);
    }
    __syncthreads();
    for (int c = 0; c < 32; ++c) { smi[n][c] = t38[n][c]; smi[n][32 + c] = gsum[c]; }

    // ---- attention scores ----
    for (int o = 0; o < 64; ++o) {
        float acc = b_a1[o];
        for (int j = 0; j < 64; ++j) acc += w_a1[o * 64 + j] * smi[n][j];
        t64[n][o] = acc > 0.f ? acc : 0.f;
    }
    for (int o = 0; o < 32; ++o) {
        float acc = b_a2[o];
        for (int j = 0; j < 64; ++j) acc += w_a2[o * 64 + j] * t64[n][j];
        t38[n][o] = acc > 0.f ? acc : 0.f;
    }
    float score = b_a3[0];
    for (int j = 0; j < 32; ++j) score += w_a3[j] * t38[n][j];
    sebuf[n] = (score != 0.f) ? __expf(score) : 0.f;
    __syncthreads();
    if (n == 0) {
        float s = 0.f;
        for (int j = 0; j < 128; ++j) s += sebuf[j];
        ssum = s;
    }
    __syncthreads();
    float wn = sebuf[n] / ssum;
    for (int c = 0; c < 64; ++c) wgt[n][c] = wn * smi[n][c];
    __syncthreads();

    // ---- GRU column n (hx = [h0 ; gru_input], raw-reshape aliasing) ----
    int d = n >> 6, mc = n & 63;                    // gru_input[i][n] = wgt[2i+d][mc]
    for (int o = 0; o < 64; ++o) {                  // pass 1: reset gate r
        float acc = 0.f;
        for (int c = 0; c < 64; ++c) acc += w_r[o * 128 + c] * smi[n][c];
        for (int i = 0; i < 64; ++i) acc += w_r[o * 128 + 64 + i] * wgt[2 * i + d][mc];
        t64[n][o] = 1.f / (1.f + __expf(-acc));
    }
    for (int o = 0; o < 64; ++o) {                  // pass 2: z, q, hn
        float az = 0.f, aq = 0.f;
        for (int c = 0; c < 64; ++c) {
            az += w_z[o * 128 + c] * smi[n][c];
            aq += w_q[o * 128 + c] * (t64[n][c] * smi[n][c]);
        }
        for (int i = 0; i < 64; ++i) {
            float gi = wgt[2 * i + d][mc];
            az += w_z[o * 128 + 64 + i] * gi;
            aq += w_q[o * 128 + 64 + i] * gi;
        }
        float z = 1.f / (1.f + __expf(-az));
        float q = tanhf(aq);
        hnm[n][o] = (1.f - z) * smi[n][o] + z * q;
    }
    __syncthreads();
    if (n < 64) {                                   // hn = mean over columns
        float s = 0.f;
        for (int j = 0; j < 128; ++j) s += hnm[j][n];
        hnacc[n] = s * (1.f / 128.f);
    }
    __syncthreads();

    // ---- head: joint(70) -> 128 -> 64 -> 1 ----
    {
        float acc = b_m1[n];
        for (int j = 0; j < 6; ++j)  acc += w_m1[n * 70 + j] * state[b * 128 * 13 + j];
        for (int j = 0; j < 64; ++j) acc += w_m1[n * 70 + 6 + j] * hnacc[j];
        m1b[n] = acc > 0.f ? acc : 0.f;
    }
    __syncthreads();
    if (n < 64) {
        float acc = b_m2[n];
        for (int j = 0; j < 128; ++j) acc += w_m2[n * 128 + j] * m1b[j];
        m2b[n] = acc > 0.f ? acc : 0.f;
    }
    __syncthreads();
    if (n == 0) {
        float acc = b_m3[0];
        for (int j = 0; j < 64; ++j) acc += w_m3[j] * m2b[j];
        out[b] = acc;
    }
}

// ---------------------------------------------------------------------------
extern "C" void kernel_launch(void* const* d_in, const int* in_sizes, int n_in,
                              void* d_out, int out_size, void* d_ws, size_t ws_size,
                              hipStream_t stream)
{
    const float* state = (const float*)d_in[0];
    const float* w_in1 = (const float*)d_in[1];  const float* b_in1 = (const float*)d_in[2];
    const float* w_in2 = (const float*)d_in[3];  const float* b_in2 = (const float*)d_in[4];
    const float* w_c1  = (const float*)d_in[5];  const float* b_c1  = (const float*)d_in[6];
    const float* w_c2  = (const float*)d_in[7];  const float* b_c2  = (const float*)d_in[8];
    const float* w_ia1 = (const float*)d_in[9];  const float* b_ia1 = (const float*)d_in[10];
    const float* w_ia2 = (const float*)d_in[11]; const float* b_ia2 = (const float*)d_in[12];
    const float* w_s1  = (const float*)d_in[13]; const float* b_s1  = (const float*)d_in[14];
    const float* w_s2  = (const float*)d_in[15]; const float* b_s2  = (const float*)d_in[16];
    const float* w_a1  = (const float*)d_in[17]; const float* b_a1  = (const float*)d_in[18];
    const float* w_a2  = (const float*)d_in[19]; const float* b_a2  = (const float*)d_in[20];
    const float* w_a3  = (const float*)d_in[21]; const float* b_a3  = (const float*)d_in[22];
    const float* w_z   = (const float*)d_in[23];
    const float* w_r   = (const float*)d_in[24];
    const float* w_q   = (const float*)d_in[25];
    const float* w_m1  = (const float*)d_in[26]; const float* b_m1  = (const float*)d_in[27];
    const float* w_m2  = (const float*)d_in[28]; const float* b_m2  = (const float*)d_in[29];
    const float* w_m3  = (const float*)d_in[30]; const float* b_m3  = (const float*)d_in[31];

    char* ws = (char*)d_ws;
    float* in_out = (float*)(ws);                        // 1 MiB
    float* fT     = (float*)(ws + (1ul  << 20));         // 1 MiB
    float* afa    = (float*)(ws + (2ul  << 20));         // 1 MiB
    h16*   wpack1 = (h16*)  (ws + (3ul  << 20));         // 36 KiB
    h16*   wpack2 = (h16*)  (ws + (3ul  << 20) + (1ul << 16));
    h16*   pairP  = (h16*)  (ws + (4ul  << 20));         // 66 MiB (padded)
    h16*   c1bufP = (h16*)  (ws + (80ul << 20));         // 132 MiB (padded)

    k_prep<<<144, 256, 0, stream>>>(w_c1, w_c2, wpack1, wpack2);
    k_front<<<32, 256, 0, stream>>>(state, w_in1, b_in1, w_in2, b_in2, in_out, fT);
    k_pair<<<262144, 128, 0, stream>>>(fT, pairP);
    k_conv1<<<8192, 256, 0, stream>>>(pairP, wpack1, b_c1, c1bufP);
    k_conv2_softmax_afa<<<8192, 256, 0, stream>>>(c1bufP, pairP, wpack2, b_c2, afa);
    k_tail<<<64, 128, 0, stream>>>(state, in_out, afa,
        w_ia1, b_ia1, w_ia2, b_ia2, w_s1, b_s1, w_s2, b_s2,
        w_a1, b_a1, w_a2, b_a2, w_a3, b_a3, w_z, w_r, w_q,
        w_m1, b_m1, w_m2, b_m2, w_m3, b_m3, (float*)d_out);
}
